// LFADS_72541997629852
// MI455X (gfx1250) — matrix operand order repaired and checked by reading, not compiled, not run on hardware
//
#include <hip/hip_runtime.h>

// ---------------------------------------------------------------------------
// LFADS for MI455X (gfx1250): bf16 WMMA GEMMs + f32 pointwise GRU gates.
// ---------------------------------------------------------------------------

typedef __bf16 bf16_t;
typedef __attribute__((ext_vector_type(16))) __bf16 v16bf;
typedef __attribute__((ext_vector_type(8)))  float  v8f;

constexpr int T_   = 200;
constexpr int B_   = 512;
constexpr int N_   = 256;
constexpr int ENC_ = 256;            // ENC_G == ENC_C == 256
constexpr int CD_  = 128;
constexpr int GD_  = 256;
constexpr int UD_  = 64;
constexpr int FD_  = 64;
constexpr int CIN_ = 2 * ENC_ + FD_; // 576
constexpr int EW_  = 3 * ENC_;       // 768

// output layout (floats), in reference return order
constexpr long long O_MEAN = 0;
constexpr long long O_LOGV = (long long)B_ * GD_;
constexpr long long O_UM   = 2LL * B_ * GD_;
constexpr long long O_UL   = O_UM + (long long)B_ * T_ * UD_;
constexpr long long O_FACT = O_UL + (long long)B_ * T_ * UD_;
constexpr long long O_RATE = O_FACT + (long long)B_ * T_ * FD_;

// ---------------------------------------------------------------------------
// WMMA tile loaders (bf16, 16x16x32).
// A layout: lane L -> row m=L&15; K elems contiguous h*16 .. h*16+15
// B layout: lane L -> col n=L&15; K elems contiguous h*16 .. h*16+15
// ---------------------------------------------------------------------------
__device__ __forceinline__ v16bf ld_a_tile(const bf16_t* p) {
  union { uint4 q[2]; v16bf v; } u;
  u.q[0] = *reinterpret_cast<const uint4*>(p);
  u.q[1] = *reinterpret_cast<const uint4*>(p + 8);
  return u.v;
}
__device__ __forceinline__ v16bf ld_b_tile(const bf16_t* p) {
  union { uint4 q[2]; v16bf v; } u;
  u.q[0] = *reinterpret_cast<const uint4*>(p);
  u.q[1] = *reinterpret_cast<const uint4*>(p + 8);
  return u.v;
}

// ---------------------------------------------------------------------------
// C[M,N] = A[M,K] @ W[N,K]^T + bias[N]   (z-batched via strides)
// block = 128 threads (4 waves); block tile 64(M) x 64(N); wave tile 16 x 64.
// Requires: N % 64 == 0, K % 32 == 0, 16B-aligned bf16 buffers, K % 8 == 0.
// ---------------------------------------------------------------------------
__global__ __launch_bounds__(128)
void k_gemm(const bf16_t* __restrict__ A, long long sA,
            const bf16_t* __restrict__ W, long long sW,
            const float*  __restrict__ bias, long long sB,
            float* __restrict__ C, long long sC,
            int M, int N, int K, int ldc)
{
  const int z = blockIdx.z;
  A    += (long long)z * sA;
  W    += (long long)z * sW;
  bias += (long long)z * sB;
  C    += (long long)z * sC;

  const int wave = threadIdx.x >> 5;
  const int lane = threadIdx.x & 31;
  const int m0 = (blockIdx.y << 6) + (wave << 4);
  const int n0 = blockIdx.x << 6;
  if (m0 >= M) return;                       // wave-uniform: EXEC stays all-1

  const int sub = lane & 15;
  const int hi  = lane >> 4;

  const bf16_t* aP  = A + (long long)(m0 + sub) * K + (hi << 4);
  const bf16_t* bP0 = W + (long long)(n0 +  0 + sub) * K + (hi << 4);
  const bf16_t* bP1 = W + (long long)(n0 + 16 + sub) * K + (hi << 4);
  const bf16_t* bP2 = W + (long long)(n0 + 32 + sub) * K + (hi << 4);
  const bf16_t* bP3 = W + (long long)(n0 + 48 + sub) * K + (hi << 4);

  v8f acc[4] = {};
  for (int k = 0; k < K; k += 32) {
    v16bf a  = ld_a_tile(aP  + k);
    v16bf b0 = ld_b_tile(bP0 + k);
    v16bf b1 = ld_b_tile(bP1 + k);
    v16bf b2 = ld_b_tile(bP2 + k);
    v16bf b3 = ld_b_tile(bP3 + k);
    acc[0] = __builtin_amdgcn_wmma_f32_16x16x32_bf16(false, a, false, b0, (short)0, acc[0], false, false);
    acc[1] = __builtin_amdgcn_wmma_f32_16x16x32_bf16(false, a, false, b1, (short)0, acc[1], false, false);
    acc[2] = __builtin_amdgcn_wmma_f32_16x16x32_bf16(false, a, false, b2, (short)0, acc[2], false, false);
    acc[3] = __builtin_amdgcn_wmma_f32_16x16x32_bf16(false, a, false, b3, (short)0, acc[3], false, false);
  }

  const int mb = m0 + (hi << 3);             // C/D layout: VGPR r -> row mb+r
  #pragma unroll
  for (int j = 0; j < 4; ++j) {
    const int n = n0 + (j << 4) + sub;
    const float bv = bias[n];
    #pragma unroll
    for (int r = 0; r < 8; ++r)
      C[(long long)(mb + r) * ldc + n] = acc[j][r] + bv;
  }
}

// ---------------------------------------------------------------------------
// Pointwise kernels
// ---------------------------------------------------------------------------
__device__ __forceinline__ float sigmoidf_(float x) { return 1.f / (1.f + expf(-x)); }

__global__ __launch_bounds__(256)
void k_cvt(const float* __restrict__ s, bf16_t* __restrict__ d, long long n) {
  long long i = (long long)blockIdx.x * blockDim.x + threadIdx.x;
  if (i < n) d[i] = (bf16_t)s[i];
}

// x[B,T,N] f32 -> xT[T,B,N] bf16
__global__ __launch_bounds__(256)
void k_xT(const float* __restrict__ x, bf16_t* __restrict__ xT) {
  long long i = (long long)blockIdx.x * blockDim.x + threadIdx.x;
  if (i >= (long long)B_ * T_ * N_) return;
  long long b = i / ((long long)T_ * N_);
  long long r = i - b * T_ * N_;
  long long t = r / N_;
  long long n = r - t * N_;
  xT[t * B_ * N_ + b * N_ + n] = (bf16_t)x[i];
}

// encoder gate kernel, z-batched over {0:gf,1:cf,2:gb,3:cb}; clamp(max=5)
// z==1 writes cin[t][:,0:256]; z==3 writes cin[T-1-t][:,256:512]
__global__ __launch_bounds__(256)
void k_enc_point(const float* __restrict__ gi, const float* __restrict__ gh,
                 float* __restrict__ h32, bf16_t* __restrict__ h16,
                 bf16_t* __restrict__ cin, int t)
{
  int i = blockIdx.x * blockDim.x + threadIdx.x;
  if (i >= 4 * B_ * ENC_) return;
  int z = i / (B_ * ENC_);
  int r = i - z * B_ * ENC_;
  int b = r / ENC_;
  int j = r - b * ENC_;
  const float* gib = gi + (long long)z * B_ * EW_ + (long long)b * EW_;
  const float* ghb = gh + (long long)z * B_ * EW_ + (long long)b * EW_;
  float rg = sigmoidf_(gib[j] + ghb[j]);
  float zg = sigmoidf_(gib[ENC_ + j] + ghb[ENC_ + j]);
  float ng = tanhf(gib[2 * ENC_ + j] + rg * ghb[2 * ENC_ + j]);
  float h  = h32[i];
  float hn = fminf((1.f - zg) * ng + zg * h, 5.0f);
  h32[i] = hn;
  h16[i] = (bf16_t)hn;
  if (z == 1)
    cin[(long long)t * B_ * CIN_ + (long long)b * CIN_ + j] = (bf16_t)hn;
  else if (z == 3)
    cin[(long long)(T_ - 1 - t) * B_ * CIN_ + (long long)b * CIN_ + ENC_ + j] = (bf16_t)hn;
}

// generator GRU gate kernel, clip(0,5)
__global__ __launch_bounds__(256)
void k_gru_point(const float* __restrict__ gi, const float* __restrict__ gh,
                 float* __restrict__ h32, bf16_t* __restrict__ h16, int H)
{
  int i = blockIdx.x * blockDim.x + threadIdx.x;
  if (i >= B_ * H) return;
  int b = i / H;
  int j = i - b * H;
  const float* gib = gi + (long long)b * 3 * H;
  const float* ghb = gh + (long long)b * 3 * H;
  float rg = sigmoidf_(gib[j] + ghb[j]);
  float zg = sigmoidf_(gib[H + j] + ghb[H + j]);
  float ng = tanhf(gib[2 * H + j] + rg * ghb[2 * H + j]);
  float h  = h32[i];
  float hn = fminf(fmaxf((1.f - zg) * ng + zg * h, 0.0f), 5.0f);
  h32[i] = hn;
  h16[i] = (bf16_t)hn;
}

// heg = concat(hgf(z0), hgb(z2)) in bf16
__global__ __launch_bounds__(256)
void k_concat_heg(const bf16_t* __restrict__ h16, bf16_t* __restrict__ heg) {
  int i = blockIdx.x * blockDim.x + threadIdx.x;
  if (i >= B_ * 2 * ENC_) return;
  int b = i / (2 * ENC_);
  int j = i - b * 2 * ENC_;
  heg[i] = (j < ENC_) ? h16[(long long)0 * B_ * ENC_ + (long long)b * ENC_ + j]
                      : h16[(long long)2 * B_ * ENC_ + (long long)b * ENC_ + (j - ENC_)];
}

// g0_logv = log(exp(raw)+1e-4) in place; g0 = eps*exp(0.5*logv)+mean
__global__ __launch_bounds__(256)
void k_g0_point(const float* __restrict__ mean, float* __restrict__ logv,
                const float* __restrict__ eps, float* __restrict__ g32,
                bf16_t* __restrict__ g16)
{
  int i = blockIdx.x * blockDim.x + threadIdx.x;
  if (i >= B_ * GD_) return;
  float v = logf(expf(logv[i]) + 1e-4f);
  logv[i] = v;
  float g = eps[i] * expf(0.5f * v) + mean[i];
  g32[i] = g;
  g16[i] = (bf16_t)g;
}

// u = eps_u * exp(0.5*ul) + um   (um/ul already in d_out at [b,t,:])
__global__ __launch_bounds__(256)
void k_u_point(const float* __restrict__ um, const float* __restrict__ ul,
               const float* __restrict__ eps_u, bf16_t* __restrict__ u16, int t)
{
  int i = blockIdx.x * blockDim.x + threadIdx.x;
  if (i >= B_ * UD_) return;
  int b = i / UD_;
  int j = i - b * UD_;
  long long idx = (long long)b * T_ * UD_ + (long long)t * UD_ + j;
  float u = eps_u[idx] * expf(0.5f * ul[idx]) + um[idx];
  u16[i] = (bf16_t)u;
}

// stash factor f (already in d_out) as bf16 for rates GEMM + next cin slot
__global__ __launch_bounds__(256)
void k_f_point(const float* __restrict__ fac, bf16_t* __restrict__ fact16,
               bf16_t* __restrict__ cin, int t)
{
  int i = blockIdx.x * blockDim.x + threadIdx.x;
  if (i >= B_ * FD_) return;
  int b = i / FD_;
  int j = i - b * FD_;
  long long idx = (long long)b * T_ * FD_ + (long long)t * FD_ + j;
  float f = fac[idx];
  fact16[idx] = (bf16_t)f;
  if (t + 1 < T_)
    cin[(long long)(t + 1) * B_ * CIN_ + (long long)b * CIN_ + 2 * ENC_ + j] = (bf16_t)f;
}

// f0 -> cin[0][:,512:576]
__global__ __launch_bounds__(256)
void k_f0_point(const float* __restrict__ f0, bf16_t* __restrict__ cin) {
  int i = blockIdx.x * blockDim.x + threadIdx.x;
  if (i >= B_ * FD_) return;
  int b = i / FD_;
  int j = i - b * FD_;
  cin[(long long)b * CIN_ + 2 * ENC_ + j] = (bf16_t)f0[i];
}

// rates = exp(clip(x,-5,5)) in place
__global__ __launch_bounds__(256)
void k_rates(float* __restrict__ p, long long n) {
  long long i = (long long)blockIdx.x * blockDim.x + threadIdx.x;
  if (i < n) p[i] = expf(fminf(fmaxf(p[i], -5.0f), 5.0f));
}

// ---------------------------------------------------------------------------
// Host orchestration
// ---------------------------------------------------------------------------
extern "C" void kernel_launch(void* const* d_in, const int* in_sizes, int n_in,
                              void* d_out, int out_size, void* d_ws, size_t ws_size,
                              hipStream_t stream)
{
  (void)in_sizes; (void)n_in; (void)out_size; (void)ws_size;
  float* out = (float*)d_out;
  auto in = [&](int i) { return (const float*)d_in[i]; };

  // d_in order: 0 x, 1 eps_g0, 2 eps_u,
  //   gf:3-6  gb:7-10  cf:11-14  cb:15-18  ct:19-22  gn:23-26   (Wih,Whh,bih,bhh)
  //   g0m:27,28  g0v:29,30  um:31,32  uv:33,34  f:35,36  r:37,38 (W,b)

  char* base = (char*)d_ws;
  size_t off = 0;
  auto alloc = [&](size_t bytes) -> void* {
    void* p = base + off;
    off = (off + bytes + 255) & ~(size_t)255;
    return p;
  };

  // bf16 workspace
  bf16_t* xT     = (bf16_t*)alloc(2ull * T_ * B_ * N_);
  bf16_t* WihE   = (bf16_t*)alloc(2ull * 4 * EW_ * N_);
  bf16_t* WhhE   = (bf16_t*)alloc(2ull * 4 * EW_ * ENC_);
  bf16_t* WctI   = (bf16_t*)alloc(2ull * 3 * CD_ * CIN_);
  bf16_t* WctH   = (bf16_t*)alloc(2ull * 3 * CD_ * CD_);
  bf16_t* WgnI   = (bf16_t*)alloc(2ull * 3 * GD_ * UD_);
  bf16_t* WgnH   = (bf16_t*)alloc(2ull * 3 * GD_ * GD_);
  bf16_t* Wg0m   = (bf16_t*)alloc(2ull * GD_ * 2 * ENC_);
  bf16_t* Wg0v   = (bf16_t*)alloc(2ull * GD_ * 2 * ENC_);
  bf16_t* Wum    = (bf16_t*)alloc(2ull * UD_ * CD_);
  bf16_t* Wuv    = (bf16_t*)alloc(2ull * UD_ * CD_);
  bf16_t* Wf     = (bf16_t*)alloc(2ull * FD_ * GD_);
  bf16_t* Wr     = (bf16_t*)alloc(2ull * N_ * FD_);
  bf16_t* hE16   = (bf16_t*)alloc(2ull * 4 * B_ * ENC_);
  bf16_t* heg    = (bf16_t*)alloc(2ull * B_ * 2 * ENC_);
  bf16_t* cin    = (bf16_t*)alloc(2ull * T_ * B_ * CIN_);
  bf16_t* hc16   = (bf16_t*)alloc(2ull * B_ * CD_);
  bf16_t* g16    = (bf16_t*)alloc(2ull * B_ * GD_);
  bf16_t* u16    = (bf16_t*)alloc(2ull * B_ * UD_);
  bf16_t* fact16 = (bf16_t*)alloc(2ull * B_ * T_ * FD_);
  // f32 workspace
  float* hE32  = (float*)alloc(4ull * 4 * B_ * ENC_);
  float* biE   = (float*)alloc(4ull * 4 * EW_);
  float* bhE   = (float*)alloc(4ull * 4 * EW_);
  float* giE   = (float*)alloc(4ull * 4 * B_ * EW_);
  float* ghE   = (float*)alloc(4ull * 4 * B_ * EW_);
  float* hc32  = (float*)alloc(4ull * B_ * CD_);
  float* gi_ct = (float*)alloc(4ull * B_ * 3 * CD_);
  float* gh_ct = (float*)alloc(4ull * B_ * 3 * CD_);
  float* g32   = (float*)alloc(4ull * B_ * GD_);
  float* gi_gn = (float*)alloc(4ull * B_ * 3 * GD_);
  float* gh_gn = (float*)alloc(4ull * B_ * 3 * GD_);
  float* f0raw = (float*)alloc(4ull * B_ * FD_);

  auto CVT = [&](const float* s, bf16_t* d, long long n) {
    k_cvt<<<dim3((unsigned)((n + 255) / 256)), dim3(256), 0, stream>>>(s, d, n);
  };
  auto GEMM = [&](const bf16_t* A, long long sA, const bf16_t* W, long long sW,
                  const float* bias, long long sB, float* C, long long sC,
                  int M, int N, int K, int ldc, int zc) {
    k_gemm<<<dim3(N / 64, (M + 63) / 64, zc), dim3(128), 0, stream>>>(
        A, sA, W, sW, bias, sB, C, sC, M, N, K, ldc);
  };

  // ---- setup: conversions, bias packing, zero states --------------------
  {
    long long n = (long long)B_ * T_ * N_;
    k_xT<<<dim3((unsigned)((n + 255) / 256)), dim3(256), 0, stream>>>(in(0), xT);
  }
  // encoder z-order: 0=gf 1=cf 2=gb 3=cb
  const int eWih[4] = {3, 11, 7, 15}, eWhh[4] = {4, 12, 8, 16};
  const int eBih[4] = {5, 13, 9, 17}, eBhh[4] = {6, 14, 10, 18};
  for (int z = 0; z < 4; ++z) {
    CVT(in(eWih[z]), WihE + (long long)z * EW_ * N_,   (long long)EW_ * N_);
    CVT(in(eWhh[z]), WhhE + (long long)z * EW_ * ENC_, (long long)EW_ * ENC_);
    hipMemcpyAsync(biE + z * EW_, in(eBih[z]), sizeof(float) * EW_,
                   hipMemcpyDeviceToDevice, stream);
    hipMemcpyAsync(bhE + z * EW_, in(eBhh[z]), sizeof(float) * EW_,
                   hipMemcpyDeviceToDevice, stream);
  }
  CVT(in(19), WctI, 3LL * CD_ * CIN_);
  CVT(in(20), WctH, 3LL * CD_ * CD_);
  CVT(in(23), WgnI, 3LL * GD_ * UD_);
  CVT(in(24), WgnH, 3LL * GD_ * GD_);
  CVT(in(27), Wg0m, (long long)GD_ * 2 * ENC_);
  CVT(in(29), Wg0v, (long long)GD_ * 2 * ENC_);
  CVT(in(31), Wum,  (long long)UD_ * CD_);
  CVT(in(33), Wuv,  (long long)UD_ * CD_);
  CVT(in(35), Wf,   (long long)FD_ * GD_);
  CVT(in(37), Wr,   (long long)N_ * FD_);

  hipMemsetAsync(hE32, 0, 4ull * 4 * B_ * ENC_, stream);
  hipMemsetAsync(hE16, 0, 2ull * 4 * B_ * ENC_, stream);
  hipMemsetAsync(hc32, 0, 4ull * B_ * CD_, stream);
  hipMemsetAsync(hc16, 0, 2ull * B_ * CD_, stream);

  // ---- encoder: 4 recurrent chains, 200 steps ---------------------------
  const long long sWihZ = (long long)EW_ * N_;
  const long long sWhhZ = (long long)EW_ * ENC_;
  const long long sGz   = (long long)B_ * EW_;
  for (int t = 0; t < T_; ++t) {
    const bf16_t* xf = xT + (long long)t * B_ * N_;
    const bf16_t* xb = xT + (long long)(T_ - 1 - t) * B_ * N_;
    // gi = x @ Wih^T + bih : forward pair (z0,z1), backward pair (z2,z3)
    GEMM(xf, 0, WihE, sWihZ, biE, EW_, giE, sGz, B_, EW_, N_, EW_, 2);
    GEMM(xb, 0, WihE + 2 * sWihZ, sWihZ, biE + 2 * EW_, EW_,
         giE + 2 * sGz, sGz, B_, EW_, N_, EW_, 2);
    // gh = h @ Whh^T + bhh : all four chains
    GEMM(hE16, (long long)B_ * ENC_, WhhE, sWhhZ, bhE, EW_, ghE, sGz,
         B_, EW_, ENC_, EW_, 4);
    k_enc_point<<<dim3((4 * B_ * ENC_ + 255) / 256), dim3(256), 0, stream>>>(
        giE, ghE, hE32, hE16, cin, t);
  }

  // ---- g0 sampling, f0 --------------------------------------------------
  k_concat_heg<<<dim3((B_ * 2 * ENC_ + 255) / 256), dim3(256), 0, stream>>>(hE16, heg);
  GEMM(heg, 0, Wg0m, 0, in(28), 0, out + O_MEAN, 0, B_, GD_, 2 * ENC_, GD_, 1);
  GEMM(heg, 0, Wg0v, 0, in(30), 0, out + O_LOGV, 0, B_, GD_, 2 * ENC_, GD_, 1);
  k_g0_point<<<dim3((B_ * GD_ + 255) / 256), dim3(256), 0, stream>>>(
      out + O_MEAN, out + O_LOGV, in(1), g32, g16);
  GEMM(g16, 0, Wf, 0, in(36), 0, f0raw, 0, B_, FD_, GD_, FD_, 1);
  k_f0_point<<<dim3((B_ * FD_ + 255) / 256), dim3(256), 0, stream>>>(f0raw, cin);

  // ---- generator: 200 sequential steps ----------------------------------
  for (int t = 0; t < T_; ++t) {
    const bf16_t* cint = cin + (long long)t * B_ * CIN_;
    // controller GRU (576 -> 128), clip(0,5)
    GEMM(cint, 0, WctI, 0, in(21), 0, gi_ct, 0, B_, 3 * CD_, CIN_, 3 * CD_, 1);
    GEMM(hc16, 0, WctH, 0, in(22), 0, gh_ct, 0, B_, 3 * CD_, CD_, 3 * CD_, 1);
    k_gru_point<<<dim3((B_ * CD_ + 255) / 256), dim3(256), 0, stream>>>(
        gi_ct, gh_ct, hc32, hc16, CD_);
    // u mean / logvar straight into d_out, then sample u
    GEMM(hc16, 0, Wum, 0, in(32), 0, out + O_UM + (long long)t * UD_, 0,
         B_, UD_, CD_, T_ * UD_, 1);
    GEMM(hc16, 0, Wuv, 0, in(34), 0, out + O_UL + (long long)t * UD_, 0,
         B_, UD_, CD_, T_ * UD_, 1);
    k_u_point<<<dim3((B_ * UD_ + 255) / 256), dim3(256), 0, stream>>>(
        out + O_UM, out + O_UL, in(2), u16, t);
    // generator GRU (64 -> 256), clip(0,5)
    GEMM(u16, 0, WgnI, 0, in(25), 0, gi_gn, 0, B_, 3 * GD_, UD_, 3 * GD_, 1);
    GEMM(g16, 0, WgnH, 0, in(26), 0, gh_gn, 0, B_, 3 * GD_, GD_, 3 * GD_, 1);
    k_gru_point<<<dim3((B_ * GD_ + 255) / 256), dim3(256), 0, stream>>>(
        gi_gn, gh_gn, g32, g16, GD_);
    // factors straight into d_out; stash bf16 + feed cin[t+1]
    GEMM(g16, 0, Wf, 0, in(36), 0, out + O_FACT + (long long)t * FD_, 0,
         B_, FD_, GD_, T_ * FD_, 1);
    k_f_point<<<dim3((B_ * FD_ + 255) / 256), dim3(256), 0, stream>>>(
        out + O_FACT, fact16, cin, t);
  }

  // ---- rates: one big parallel GEMM + pointwise exp(clip) ----------------
  GEMM(fact16, 0, Wr, 0, in(38), 0, out + O_RATE, 0,
       B_ * T_, N_, FD_, N_, 1);
  {
    long long n = (long long)B_ * T_ * N_;
    k_rates<<<dim3((unsigned)((n + 255) / 256)), dim3(256), 0, stream>>>(out + O_RATE, n);
  }
}
